// INN_TTS_44976897523929
// MI455X (gfx1250) — matrix operand order repaired and checked by reading, not compiled, not run on hardware
//
#include <hip/hip_runtime.h>
#include <cstdint>
#include <cstddef>

// ---------------------------------------------------------------------------
// Model constants (from reference)
// ---------------------------------------------------------------------------
#define B_SZ    8
#define T_TXT   256
#define S_AUD   256
#define DM      512      // D_MODEL
#define H_LSTM  256
#define N_NEUR  32
#define NH      4
#define DH      128
#define V_AUD   4096

typedef __bf16 bf16_t;
typedef __attribute__((ext_vector_type(16))) __bf16 v16bf;
typedef __attribute__((ext_vector_type(8)))  float  v8f;
typedef __attribute__((ext_vector_type(4)))  unsigned u32x4;

union FragBF { v16bf v; u32x4 q[2]; bf16_t e[16]; };
union FragF  { v8f   v; float  e[8];  };

__device__ __forceinline__ bf16_t f2bf(float f) {
  union { unsigned short u; bf16_t b; } cv;
  unsigned x = __float_as_uint(f);
  cv.u = (unsigned short)((x + 0x7FFFu + ((x >> 16) & 1u)) >> 16);  // RNE
  return cv.b;
}
__device__ __forceinline__ float sigf(float x) { return 1.0f / (1.0f + expf(-x)); }

// ---------------------------------------------------------------------------
// WMMA fragment loaders (16x16x32 bf16, wave32; m/n/D per CDNA5 ISA 7.12.2, k order contiguous in A and B)
// A (16x32): lanes 0-15 rows M, elems0-15=K0..15; lanes16-31: K16..31 (same k order as B)
// B (32x16): lane%16 = N column, lanes<16 K=k0..k0+15, lanes>=16 K=k0+16..31
// D (16x16 f32): VGPR r -> M = r + 8*(lane/16), N = lane%16
// Padding rows are CLAMPED (not zeroed): D row m depends only on A row m, and
// out-of-range rows are never stored, so no EXEC juggling in the hot loop.
// All addresses are 16B aligned (lda/ldw multiples of 8 bf16, k0 mult of 32).
// ---------------------------------------------------------------------------
__device__ __forceinline__ v16bf loadA_frag(const bf16_t* A, int lda, int M,
                                            int mBase, int k0, int lane) {
  int m = mBase + (lane & 15);
  if (m >= M) m = M - 1;                       // clamp padding rows
  int hi = lane >> 4;
  const bf16_t* p = A + (size_t)m * lda + k0 + hi * 16;
  FragBF f;
  f.q[0] = *(const u32x4*)(p);                 // K = k0+hi*16 .. +7
  f.q[1] = *(const u32x4*)(p + 8);             // K = k0+hi*16+8 .. +15
  return f.v;
}

__device__ __forceinline__ v16bf loadB_frag(const bf16_t* W, int ldw,
                                            int nBase, int k0, int lane) {
  int n = nBase + (lane & 15);
  const bf16_t* p = W + (size_t)n * ldw + k0 + ((lane >> 4) << 4);
  __builtin_prefetch((const void*)(p + 256), 0, 1);   // global_prefetch_b8
  FragBF f;
  f.q[0] = *(const u32x4*)(p);
  f.q[1] = *(const u32x4*)(p + 8);
  return f.v;
}

// ---------------------------------------------------------------------------
// Generic batched GEMM:  C[z] = A[z] @ W[z]^T (+bias) (+residual)
// A: [M,K] bf16 (row stride lda), W: [N,K] bf16 (row stride ldw)
// Each wave computes a 16 x (16*NT) strip: A fragment loaded once per k-step,
// reused across NT B-tiles (4 independent back-to-back WMMAs for the XDL pipe).
// grid = (N/(16*NT), ceil(M/16), batches), block = 32 (one wave)
// ---------------------------------------------------------------------------
#define NT 4
__global__ __launch_bounds__(32) void
gemm_bf16_wmma(const bf16_t* __restrict__ A, long long sA, int lda,
               const bf16_t* __restrict__ W, long long sW, int ldw,
               float* __restrict__ C, long long sC, int ldc,
               const float* __restrict__ bias, long long sBias,
               const float* __restrict__ res, long long sRes, int ldres,
               bf16_t* __restrict__ Cbf, long long sCbf, int ldcbf,
               int M, int N, int K) {
  int lane  = threadIdx.x;
  int nBase = blockIdx.x * (16 * NT);
  int mBase = blockIdx.y << 4;
  int z     = blockIdx.z;
  A += (size_t)z * sA;
  W += (size_t)z * sW;
  C += (size_t)z * sC;

  FragF acc[NT];
#pragma unroll
  for (int j = 0; j < NT; ++j)
#pragma unroll
    for (int r = 0; r < 8; ++r) acc[j].e[r] = 0.0f;

  for (int k0 = 0; k0 < K; k0 += 32) {
    v16bf a = loadA_frag(A, lda, M, mBase, k0, lane);
#pragma unroll
    for (int j = 0; j < NT; ++j) {
      v16bf b = loadB_frag(W, ldw, nBase + 16 * j, k0, lane);
      acc[j].v = __builtin_amdgcn_wmma_f32_16x16x32_bf16(false, a, false, b,
                                                         (short)0, acc[j].v,
                                                         false, false);
    }
  }

#pragma unroll
  for (int j = 0; j < NT; ++j) {
    int n = nBase + 16 * j + (lane & 15);
    float bv = bias ? bias[(size_t)z * sBias + n] : 0.0f;
#pragma unroll
    for (int r = 0; r < 8; ++r) {
      int m = mBase + ((lane >> 4) << 3) + r;
      if (m < M) {
        float v = acc[j].e[r] + bv;
        if (res) v += res[(size_t)z * sRes + (size_t)m * ldres + n];
        C[(size_t)m * ldc + n] = v;
        if (Cbf) Cbf[(size_t)z * sCbf + (size_t)m * ldcbf + n] = f2bf(v);
      }
    }
  }
}

// ---------------------------------------------------------------------------
// Cross attention for one (b, head): Q [32,128] over K/V [256,128]
// grid = B*NH = 32, block = 128 (4 waves)
// ---------------------------------------------------------------------------
__global__ void cross_attn_kernel(const bf16_t* __restrict__ Qbf,   // [B*32, 512]
                                  const bf16_t* __restrict__ Kbf,   // [B*256,512]
                                  const bf16_t* __restrict__ Vbf,   // [B*256,512]
                                  float* __restrict__ ctx,          // [B*32, 512]
                                  bf16_t* __restrict__ ctxbf) {
  __shared__ bf16_t Qs[N_NEUR * DH];       // 8 KB
  __shared__ float  Sc[N_NEUR * T_TXT];    // 32 KB
  __shared__ bf16_t Ps[N_NEUR * T_TXT];    // 16 KB
  int tid  = threadIdx.x;
  int lane = tid & 31;
  int wave = tid >> 5;
  int b  = blockIdx.x >> 2;
  int hd = blockIdx.x & 3;

  for (int i = tid; i < N_NEUR * DH; i += 128) {
    int q = i >> 7, d = i & 127;
    Qs[i] = Qbf[(size_t)(b * N_NEUR + q) * DM + hd * DH + d];
  }
  __syncthreads();

  // S = Q @ K^T * 1/sqrt(128)
  const float scale = 0.08838834764831845f;
  const bf16_t* Kbase = Kbf + (size_t)b * T_TXT * DM + hd * DH;
  for (int t = wave; t < 2 * (T_TXT / 16); t += 4) {
    int mBase = (t & 1) << 4;
    int nBase = (t >> 1) << 4;
    FragF acc;
#pragma unroll
    for (int r = 0; r < 8; ++r) acc.e[r] = 0.0f;
    for (int k0 = 0; k0 < DH; k0 += 32) {
      v16bf a  = loadA_frag(Qs, DH, N_NEUR, mBase, k0, lane);
      v16bf bb = loadB_frag(Kbase, DM, nBase, k0, lane);
      acc.v = __builtin_amdgcn_wmma_f32_16x16x32_bf16(false, a, false, bb,
                                                      (short)0, acc.v, false, false);
    }
    int n = nBase + (lane & 15);
#pragma unroll
    for (int r = 0; r < 8; ++r) {
      int m = mBase + ((lane >> 4) << 3) + r;
      Sc[m * T_TXT + n] = acc.e[r] * scale;
    }
  }
  __syncthreads();

  // row softmax (32 rows handled by threads 0..31)
  if (tid < N_NEUR) {
    float mx = -3.0e38f;
    for (int kk = 0; kk < T_TXT; ++kk) mx = fmaxf(mx, Sc[tid * T_TXT + kk]);
    float den = 0.0f;
    for (int kk = 0; kk < T_TXT; ++kk) den += expf(Sc[tid * T_TXT + kk] - mx);
    float inv = 1.0f / den;
    for (int kk = 0; kk < T_TXT; ++kk)
      Ps[tid * T_TXT + kk] = f2bf(expf(Sc[tid * T_TXT + kk] - mx) * inv);
  }
  __syncthreads();

  // O = P @ V  (V read strided: B-frag element (k,n) = V[k][n])
  for (int t = wave; t < 2 * (DH / 16); t += 4) {
    int mBase = (t & 1) << 4;
    int nBase = (t >> 1) << 4;
    FragF acc;
#pragma unroll
    for (int r = 0; r < 8; ++r) acc.e[r] = 0.0f;
    for (int k0 = 0; k0 < T_TXT; k0 += 32) {
      v16bf a = loadA_frag(Ps, T_TXT, N_NEUR, mBase, k0, lane);
      FragBF fb;
      int n  = nBase + (lane & 15);
      int kb = k0 + ((lane >> 4) << 4);
      const bf16_t* vp = Vbf + (size_t)(b * T_TXT + kb) * DM + hd * DH + n;
#pragma unroll
      for (int h = 0; h < 16; ++h) fb.e[h] = vp[(size_t)h * DM];
      acc.v = __builtin_amdgcn_wmma_f32_16x16x32_bf16(false, a, false, fb.v,
                                                      (short)0, acc.v, false, false);
    }
    int n = nBase + (lane & 15);
#pragma unroll
    for (int r = 0; r < 8; ++r) {
      int m = mBase + ((lane >> 4) << 3) + r;
      size_t o = (size_t)(b * N_NEUR + m) * DM + hd * DH + n;
      ctx[o]   = acc.e[r];
      ctxbf[o] = f2bf(acc.e[r]);
    }
  }
}

// ---------------------------------------------------------------------------
// Sequential text LSTM (one direction per launch), 1 block x 256 threads.
// memory layout: [B][T][512], fwd -> cols 0..255, bwd -> cols 256..511
// ---------------------------------------------------------------------------
__global__ void text_lstm_kernel(const int* __restrict__ text,
                                 const float* __restrict__ emb,
                                 const float* __restrict__ Wih,
                                 const float* __restrict__ Whh,
                                 const float* __restrict__ bih,
                                 const float* __restrict__ bhh,
                                 float* __restrict__ memory, int dir) {
  __shared__ float hs[B_SZ * H_LSTM];       // 8 KB
  __shared__ float xs[B_SZ * DM];           // 16 KB
  __shared__ float gs[B_SZ * 4 * H_LSTM];   // 32 KB
  int tid = threadIdx.x;
  float creg[B_SZ];
#pragma unroll
  for (int j = 0; j < B_SZ; ++j) creg[j] = 0.0f;
  for (int i = tid; i < B_SZ * H_LSTM; i += 256) hs[i] = 0.0f;
  __syncthreads();

  for (int step = 0; step < T_TXT; ++step) {
    int t = dir ? (T_TXT - 1 - step) : step;
    for (int i = tid; i < B_SZ * DM; i += 256) {
      int b = i >> 9, d = i & 511;
      xs[i] = emb[(size_t)text[b * T_TXT + t] * DM + d];
    }
    __syncthreads();
    for (int i = tid; i < B_SZ * 4 * H_LSTM; i += 256) {
      int b = i >> 10, k = i & 1023;
      float acc = bih[k] + bhh[k];
      const float* wi = Wih + (size_t)k * DM;
      const float* xb = xs + b * DM;
      for (int d = 0; d < DM; ++d) acc += xb[d] * wi[d];
      const float* wh = Whh + (size_t)k * H_LSTM;
      const float* hb = hs + b * H_LSTM;
      for (int r = 0; r < H_LSTM; ++r) acc += hb[r] * wh[r];
      gs[i] = acc;
    }
    __syncthreads();
    int r = tid;  // 0..255
#pragma unroll
    for (int b = 0; b < B_SZ; ++b) {
      float ii = sigf(gs[b * 1024 + r]);
      float ff = sigf(gs[b * 1024 + 256 + r]);
      float gg = tanhf(gs[b * 1024 + 512 + r]);
      float oo = sigf(gs[b * 1024 + 768 + r]);
      float c = ff * creg[b] + ii * gg;
      creg[b] = c;
      float hh = oo * tanhf(c);
      hs[b * H_LSTM + r] = hh;
      memory[(size_t)(b * T_TXT + t) * DM + dir * H_LSTM + r] = hh;
    }
    __syncthreads();
  }
}

// ---------------------------------------------------------------------------
// Small elementwise kernels
// ---------------------------------------------------------------------------
__global__ void pack_cw_kernel(const float* __restrict__ cWih,
                               const float* __restrict__ cWhh,
                               bf16_t* __restrict__ out) {
  long long idx = (long long)blockIdx.x * blockDim.x + threadIdx.x;  // 32*2048*1024
  int d = (int)(idx & 1023);
  long long nk = idx >> 10;                       // n*2048 + k
  float v = (d < 512) ? cWih[nk * 512 + d] : cWhh[nk * 512 + (d - 512)];
  out[idx] = f2bf(v);
}

__global__ void add_bias_kernel(const float* a, const float* b, float* o, int n) {
  int i = blockIdx.x * 256 + threadIdx.x;
  if (i < n) o[i] = a[i] + b[i];
}

__global__ void cvt_f32_bf16(const float* __restrict__ s, bf16_t* __restrict__ d, int n) {
  for (int i = blockIdx.x * 256 + threadIdx.x; i < n; i += gridDim.x * 256)
    d[i] = f2bf(s[i]);
}

__global__ void fill_zero_u32(unsigned* p, int n) {
  for (int i = blockIdx.x * 256 + threadIdx.x; i < n; i += gridDim.x * 256) p[i] = 0u;
}

__global__ void audio_embed_kernel(const int* __restrict__ audio,
                                   const float* __restrict__ emb,
                                   float* __restrict__ xe, bf16_t* __restrict__ xebf) {
  int idx = blockIdx.x * 256 + threadIdx.x;  // S*B*512 = 1048576
  int d = idx & 511; int tb = idx >> 9; int b = tb & 7; int t = tb >> 3;
  float v = emb[(size_t)audio[b * S_AUD + t] * DM + d];
  xe[idx] = v; xebf[idx] = f2bf(v);
}

// A_gate[n][b][0:512]=x_t[b], [512:1024]=h[n][b]
__global__ void build_a_kernel(const bf16_t* __restrict__ xe_t,
                               const bf16_t* __restrict__ hxbf,
                               bf16_t* __restrict__ Ag) {
  int idx = blockIdx.x * 256 + threadIdx.x;  // N*B*1024 = 262144
  int d = idx & 1023; int nb = idx >> 10; int b = nb & 7; int n = nb >> 3;
  Ag[idx] = (d < 512) ? xe_t[b * 512 + d] : hxbf[(size_t)(n * B_SZ + b) * 512 + (d - 512)];
}

// LSTM-cell nonlinearity over gate bank G[n][b][2048] (i,f,g,o order)
__global__ void cell_kernel(const float* __restrict__ G, float* __restrict__ hx,
                            float* __restrict__ cx, bf16_t* __restrict__ hxbf,
                            float* __restrict__ stackf, bf16_t* __restrict__ stackbf) {
  int idx = blockIdx.x * 256 + threadIdx.x;  // N*B*512 = 131072
  int r = idx & 511; int nb = idx >> 9; int b = nb & 7; int n = nb >> 3;
  const float* g = G + (size_t)nb * 2048;
  float ii = sigf(g[r]);
  float ff = sigf(g[512 + r]);
  float gg = tanhf(g[1024 + r]);
  float oo = sigf(g[1536 + r]);
  float c = ff * cx[idx] + ii * gg;
  cx[idx] = c;
  float hh = oo * tanhf(c);
  hx[idx] = hh;
  hxbf[idx] = f2bf(hh);
  size_t so = (size_t)(b * N_NEUR + n) * 512 + r;
  stackf[so] = hh; stackbf[so] = f2bf(hh);
}

// Mixer attention: Lq=1 per batch; grid = B*NH, block = 128
__global__ void mixer_attn_kernel(const float* __restrict__ qm,   // [8,512]
                                  const float* __restrict__ Km,   // [B*32,512]
                                  const float* __restrict__ Vm,   // [B*32,512]
                                  float* __restrict__ ctxm,       // [8,512]
                                  bf16_t* __restrict__ ctxmbf) {
  __shared__ float sc[N_NEUR];
  __shared__ float pr[N_NEUR];
  __shared__ float mx_s, den_s;
  int tid = threadIdx.x;
  int b  = blockIdx.x >> 2;
  int hd = blockIdx.x & 3;
  if (tid < N_NEUR) {
    float s = 0.0f;
    const float* q = qm + b * DM + hd * DH;
    const float* k = Km + (size_t)(b * N_NEUR + tid) * DM + hd * DH;
    for (int d = 0; d < DH; ++d) s += q[d] * k[d];
    sc[tid] = s * 0.08838834764831845f;
  }
  __syncthreads();
  if (tid == 0) {
    float m = -3.0e38f;
    for (int n = 0; n < N_NEUR; ++n) m = fmaxf(m, sc[n]);
    float den = 0.0f;
    for (int n = 0; n < N_NEUR; ++n) den += expf(sc[n] - m);
    mx_s = m; den_s = den;
  }
  __syncthreads();
  if (tid < N_NEUR) pr[tid] = expf(sc[tid] - mx_s) / den_s;
  __syncthreads();
  float o = 0.0f;
  for (int n = 0; n < N_NEUR; ++n)
    o += pr[n] * Vm[(size_t)(b * N_NEUR + n) * DM + hd * DH + tid];
  ctxm[b * DM + hd * DH + tid] = o;
  ctxmbf[b * DM + hd * DH + tid] = f2bf(o);
}

// LayerNorm over (att + x_t), 1 block per batch row
__global__ void ln_kernel(const float* __restrict__ att, const float* __restrict__ xet,
                          const float* __restrict__ g, const float* __restrict__ bb,
                          bf16_t* __restrict__ outbf) {
  __shared__ float xv[DM];
  __shared__ float red[256];
  int b = blockIdx.x, tid = threadIdx.x;
  for (int i = tid; i < DM; i += 256) xv[i] = att[b * DM + i] + xet[b * DM + i];
  __syncthreads();
  float loc = xv[tid] + xv[tid + 256];
  red[tid] = loc; __syncthreads();
  for (int s = 128; s > 0; s >>= 1) { if (tid < s) red[tid] += red[tid + s]; __syncthreads(); }
  float mean = red[0] / (float)DM;
  __syncthreads();
  float d0 = xv[tid] - mean, d1 = xv[tid + 256] - mean;
  red[tid] = d0 * d0 + d1 * d1; __syncthreads();
  for (int s = 128; s > 0; s >>= 1) { if (tid < s) red[tid] += red[tid + s]; __syncthreads(); }
  float rstd = rsqrtf(red[0] / (float)DM + 1e-5f);
  for (int i = tid; i < DM; i += 256)
    outbf[b * DM + i] = f2bf((xv[i] - mean) * rstd * g[i] + bb[i]);
}

// ---------------------------------------------------------------------------
// Host side
// ---------------------------------------------------------------------------
extern "C" void kernel_launch(void* const* d_in, const int* in_sizes, int n_in,
                              void* d_out, int out_size, void* d_ws, size_t ws_size,
                              hipStream_t stream) {
  (void)in_sizes; (void)n_in; (void)out_size; (void)ws_size;
  const int*   text      = (const int*)d_in[0];
  const int*   audio     = (const int*)d_in[1];
  const float* text_emb  = (const float*)d_in[2];
  const float* lWih_f = (const float*)d_in[3];  const float* lWhh_f = (const float*)d_in[4];
  const float* lbih_f = (const float*)d_in[5];  const float* lbhh_f = (const float*)d_in[6];
  const float* lWih_b = (const float*)d_in[7];  const float* lWhh_b = (const float*)d_in[8];
  const float* lbih_b = (const float*)d_in[9];  const float* lbhh_b = (const float*)d_in[10];
  const float* audio_emb = (const float*)d_in[11];
  const float* cWih = (const float*)d_in[12];   const float* cWhh = (const float*)d_in[13];
  const float* cbih = (const float*)d_in[14];   const float* cbhh = (const float*)d_in[15];
  const float* attW[8]  = { (const float*)d_in[16], (const float*)d_in[18],
                            (const float*)d_in[20], (const float*)d_in[22],
                            (const float*)d_in[24], (const float*)d_in[26],
                            (const float*)d_in[28], (const float*)d_in[30] };
  const float* cabq = (const float*)d_in[17]; const float* cabk = (const float*)d_in[19];
  const float* cabv = (const float*)d_in[21]; const float* cabo = (const float*)d_in[23];
  const float* mabq = (const float*)d_in[25]; const float* mabk = (const float*)d_in[27];
  const float* mabv = (const float*)d_in[29]; const float* mabo = (const float*)d_in[31];
  const float* lng  = (const float*)d_in[32]; const float* lnb  = (const float*)d_in[33];
  const float* headW = (const float*)d_in[34]; const float* headb = (const float*)d_in[35];
  float* out = (float*)d_out;

  char* ws = (char*)d_ws;
  size_t off = 0;
  auto alloc = [&](size_t bytes) -> void* {
    void* p = ws + off; off = (off + bytes + 255) & ~(size_t)255; return p;
  };
  const size_t SL = 512 * 512;  // att weight slice (elements)

  bf16_t* cW_bf     = (bf16_t*)alloc((size_t)32 * 2048 * 1024 * 2);  // 134 MB (fits L2)
  float*  cbias     = (float*) alloc((size_t)32 * 2048 * 4);
  bf16_t* headW_bf  = (bf16_t*)alloc((size_t)V_AUD * 512 * 2);
  bf16_t* attW_bf   = (bf16_t*)alloc((size_t)8 * SL * 2);
  float*  memory    = (float*) alloc((size_t)B_SZ * T_TXT * DM * 4);
  bf16_t* memory_bf = (bf16_t*)alloc((size_t)B_SZ * T_TXT * DM * 2);
  bf16_t* Kc_bf     = (bf16_t*)alloc((size_t)B_SZ * T_TXT * DM * 2);
  bf16_t* Vc_bf     = (bf16_t*)alloc((size_t)B_SZ * T_TXT * DM * 2);
  float*  xe        = (float*) alloc((size_t)S_AUD * B_SZ * DM * 4);
  bf16_t* xe_bf     = (bf16_t*)alloc((size_t)S_AUD * B_SZ * DM * 2);
  float*  hx        = (float*) alloc((size_t)N_NEUR * B_SZ * 512 * 4);
  float*  cx        = (float*) alloc((size_t)N_NEUR * B_SZ * 512 * 4);
  bf16_t* hx_bf     = (bf16_t*)alloc((size_t)N_NEUR * B_SZ * 512 * 2);
  bf16_t* A_gate    = (bf16_t*)alloc((size_t)N_NEUR * B_SZ * 1024 * 2);
  float*  G         = (float*) alloc((size_t)N_NEUR * B_SZ * 2048 * 4);
  float*  stackf    = (float*) alloc((size_t)B_SZ * N_NEUR * 512 * 4);
  bf16_t* stack_bf  = (bf16_t*)alloc((size_t)B_SZ * N_NEUR * 512 * 2);
  float*  Qf        = (float*) alloc((size_t)256 * 512 * 4);
  bf16_t* Q_bf      = (bf16_t*)alloc((size_t)256 * 512 * 2);
  float*  ctxf      = (float*) alloc((size_t)256 * 512 * 4);
  bf16_t* ctx_bf    = (bf16_t*)alloc((size_t)256 * 512 * 2);
  float*  enrf      = (float*) alloc((size_t)256 * 512 * 4);
  bf16_t* enr_bf    = (bf16_t*)alloc((size_t)256 * 512 * 2);
  float*  Km        = (float*) alloc((size_t)256 * 512 * 4);
  float*  Vm        = (float*) alloc((size_t)256 * 512 * 4);
  float*  qm        = (float*) alloc((size_t)8 * 512 * 4);
  float*  ctxm      = (float*) alloc((size_t)8 * 512 * 4);
  bf16_t* ctxm_bf   = (bf16_t*)alloc((size_t)8 * 512 * 2);
  float*  attOut    = (float*) alloc((size_t)8 * 512 * 4);
  bf16_t* mixed_bf  = (bf16_t*)alloc((size_t)8 * 512 * 2);
  float*  tmpBig    = (float*) alloc((size_t)2048 * 512 * 4);

  auto gemm = [&](const bf16_t* A, long long sA, int lda,
                  const bf16_t* W, long long sW, int ldw,
                  float* C, long long sC, int ldc,
                  const float* bias, long long sBias,
                  const float* res, long long sRes, int ldres,
                  bf16_t* Cbf, long long sCbf, int ldcbf,
                  int M, int N, int K, int batches) {
    dim3 grid(N / (16 * NT), (M + 15) / 16, batches);
    gemm_bf16_wmma<<<grid, dim3(32), 0, stream>>>(A, sA, lda, W, sW, ldw, C, sC, ldc,
                                                  bias, sBias, res, sRes, ldres,
                                                  Cbf, sCbf, ldcbf, M, N, K);
  };

  // ---- one-time preprocessing --------------------------------------------
  pack_cw_kernel<<<262144, 256, 0, stream>>>(cWih, cWhh, cW_bf);
  add_bias_kernel<<<256, 256, 0, stream>>>(cbih, cbhh, cbias, 32 * 2048);
  cvt_f32_bf16<<<4096, 256, 0, stream>>>(headW, headW_bf, V_AUD * 512);
  for (int i = 0; i < 8; ++i)
    cvt_f32_bf16<<<1024, 256, 0, stream>>>(attW[i], attW_bf + (size_t)i * SL, (int)SL);
  audio_embed_kernel<<<4096, 256, 0, stream>>>(audio, audio_emb, xe, xe_bf);
  text_lstm_kernel<<<1, 256, 0, stream>>>(text, text_emb, lWih_f, lWhh_f, lbih_f, lbhh_f, memory, 0);
  text_lstm_kernel<<<1, 256, 0, stream>>>(text, text_emb, lWih_b, lWhh_b, lbih_b, lbhh_b, memory, 1);
  cvt_f32_bf16<<<4096, 256, 0, stream>>>(memory, memory_bf, B_SZ * T_TXT * DM);
  // Cross-attn K,V computed once (memory is fixed across decode steps)
  gemm(memory_bf, 0, 512, attW_bf + 1 * SL, 0, 512, tmpBig, 0, 512, cabk, 0,
       nullptr, 0, 0, Kc_bf, 0, 512, 2048, 512, 512, 1);
  gemm(memory_bf, 0, 512, attW_bf + 2 * SL, 0, 512, tmpBig, 0, 512, cabv, 0,
       nullptr, 0, 0, Vc_bf, 0, 512, 2048, 512, 512, 1);
  fill_zero_u32<<<512, 256, 0, stream>>>((unsigned*)hx, N_NEUR * B_SZ * 512);
  fill_zero_u32<<<512, 256, 0, stream>>>((unsigned*)cx, N_NEUR * B_SZ * 512);
  fill_zero_u32<<<256, 256, 0, stream>>>((unsigned*)hx_bf, N_NEUR * B_SZ * 512 / 2);

  // ---- sequential decoder ------------------------------------------------
  for (int t = 0; t < S_AUD; ++t) {
    const bf16_t* xeb_t = xe_bf + (size_t)t * B_SZ * DM;
    const float*  xef_t = xe + (size_t)t * B_SZ * DM;

    build_a_kernel<<<1024, 256, 0, stream>>>(xeb_t, hx_bf, A_gate);
    // gate bank: 32 batched GEMMs [8 x 1024] @ [2048 x 1024]^T  (bf16 WMMA)
    gemm(A_gate, 8 * 1024, 1024, cW_bf, 2048LL * 1024, 1024, G, 8 * 2048, 2048,
         cbias, 2048, nullptr, 0, 0, nullptr, 0, 0, 8, 2048, 1024, N_NEUR);
    cell_kernel<<<512, 256, 0, stream>>>(G, hx, cx, hx_bf, stackf, stack_bf);

    // cross attention
    gemm(stack_bf, 0, 512, attW_bf + 0 * SL, 0, 512, Qf, 0, 512, cabq, 0,
         nullptr, 0, 0, Q_bf, 0, 512, 256, 512, 512, 1);
    cross_attn_kernel<<<32, 128, 0, stream>>>(Q_bf, Kc_bf, Vc_bf, ctxf, ctx_bf);
    gemm(ctx_bf, 0, 512, attW_bf + 3 * SL, 0, 512, enrf, 0, 512, cabo, 0,
         stackf, 0, 512, enr_bf, 0, 512, 256, 512, 512, 1);   // + residual(stack)

    // mixer attention
    gemm(enr_bf, 0, 512, attW_bf + 5 * SL, 0, 512, Km, 0, 512, mabk, 0,
         nullptr, 0, 0, nullptr, 0, 0, 256, 512, 512, 1);
    gemm(enr_bf, 0, 512, attW_bf + 6 * SL, 0, 512, Vm, 0, 512, mabv, 0,
         nullptr, 0, 0, nullptr, 0, 0, 256, 512, 512, 1);
    gemm(xeb_t, 0, 512, attW_bf + 4 * SL, 0, 512, qm, 0, 512, mabq, 0,
         nullptr, 0, 0, nullptr, 0, 0, 8, 512, 512, 1);
    mixer_attn_kernel<<<32, 128, 0, stream>>>(qm, Km, Vm, ctxm, ctxm_bf);
    gemm(ctxm_bf, 0, 512, attW_bf + 7 * SL, 0, 512, attOut, 0, 512, mabo, 0,
         nullptr, 0, 0, nullptr, 0, 0, 8, 512, 512, 1);

    // LN + head (writes directly into d_out[b][t][:] via ldc = S*V)
    ln_kernel<<<8, 256, 0, stream>>>(attOut, xef_t, lng, lnb, mixed_bf);
    gemm(mixed_bf, 0, 512, headW_bf, 0, 512, out + (size_t)t * V_AUD, 0,
         S_AUD * V_AUD, headb, 0, nullptr, 0, 0, nullptr, 0, 0, 8, V_AUD, 512, 1);
  }
}
